// SchNetEncoder3D_61641370632835
// MI455X (gfx1250) — matrix operand order repaired and checked by reading, not compiled, not run on hardware
//
#include <hip/hip_runtime.h>
#include <math.h>

#define N_ATOMS 1024
#define H 256
#define LDIM 512
#define NG 50
#define NI 3
#define NB 16

typedef __attribute__((ext_vector_type(16))) __bf16 v16bf;
typedef __attribute__((ext_vector_type(8)))  __bf16 v8bf;
typedef __attribute__((ext_vector_type(8)))  float  v8f;

#define FW1_S 72    // padded K-stride for K=64 operands (rbf, fw1T)
#define FW2_S 264   // padded K-stride for K=256 operands (G, fw2T, awT)

static __device__ __forceinline__ v16bf mk16(v8bf lo, v8bf hi){
  v16bf r;
#pragma unroll
  for (int i = 0; i < 8; ++i){ r[i] = lo[i]; r[8+i] = hi[i]; }
  return r;
}

// A fragment (16x32 bf16) from row-major source; rowp = base of this lane's row (m = lane&15).
// lanes 0-15 hold K = {k..k+7, k+16..k+23}; lanes 16-31 hold K = {k+8..k+15, k+24..k+31}.
static __device__ __forceinline__ v16bf afrag(const __bf16* rowp, int kbase, int lane){
  const __bf16* p = rowp + kbase + ((lane & 16) ? 8 : 0);
  return mk16(*(const v8bf*)p, *(const v8bf*)(p + 16));
}

// B fragment (32x16 bf16) from transposed [n][k] source; np = base of this lane's n-row.
// lanes 0-15 hold K = {k..k+7, k+16..k+23}; lanes 16-31 hold K = {k+8..k+15, k+24..k+31}.
static __device__ __forceinline__ v16bf bfrag(const __bf16* np, int kbase, int lane){
  const __bf16* p = np + kbase + ((lane & 16) ? 8 : 0);
  return mk16(*(const v8bf*)p, *(const v8bf*)(p + 16));
}

static __device__ __forceinline__ v8f wmma_bf16(v16bf a, v16bf b, v8f c){
  return __builtin_amdgcn_wmma_f32_16x16x32_bf16(false, a, false, b, (short)0, c, false, false);
}

static __device__ __forceinline__ void wave_lds_fence(){
  asm volatile("s_wait_dscnt 0" ::: "memory");
}

// gfx1250 async global->LDS copy (ASYNCcnt-tracked, no VGPR round trip).
static __device__ __forceinline__ void async_ld_b128(unsigned int lds_addr, unsigned int goff,
                                                     const void* gbase){
  asm volatile("global_load_async_to_lds_b128 %0, %1, %2 offset:0"
               :: "v"(lds_addr), "v"(goff), "s"(gbase)
               : "memory");
}
static __device__ __forceinline__ void wait_async(){
  asm volatile("s_wait_asynccnt 0" ::: "memory");
}

static __device__ __forceinline__ unsigned int lds_off(const void* p){
  return (unsigned int)(uintptr_t)p;   // low 32 bits of shared aperture address = LDS offset
}

static __device__ __forceinline__ float gelu_fast(float x){
  return x * __builtin_amdgcn_rcpf(1.0f + __expf(-1.702f * x));
}
static __device__ __forceinline__ float gelu_exact(float x){
  return 0.5f * x * (1.0f + erff(x * 0.70710678118654752f));
}

// ---------------------------------------------------------------- init: h = emb[z], agg = 0
__global__ __launch_bounds__(256) void k_init(const int* __restrict__ z,
                                              const float* __restrict__ emb,
                                              float* __restrict__ h, float* __restrict__ agg){
  const int i = blockIdx.x, c = threadIdx.x;
  h[i*H + c]   = emb[z[i]*H + c];
  agg[i*H + c] = 0.0f;
}

// ------------------------------------------------- pack weights: bf16, transposed [n][k], padded stride
__global__ __launch_bounds__(128) void k_pack(const float* __restrict__ fw1, const float* __restrict__ fw2,
                                              const float* __restrict__ aw1, const float* __restrict__ aw2,
                                              __bf16* __restrict__ fw1T, __bf16* __restrict__ fw2T,
                                              __bf16* __restrict__ aw1T, __bf16* __restrict__ aw2T){
  const int l = blockIdx.x >> 8, n = blockIdx.x & 255, t = threadIdx.x;
  for (int k = t; k < FW1_S; k += 128)
    fw1T[(l*256 + n)*FW1_S + k] = (__bf16)((k < NG) ? fw1[(l*NG + k)*H + n] : 0.0f);
  for (int k = t; k < FW2_S; k += 128){
    const bool v = (k < H);
    fw2T[(l*256 + n)*FW2_S + k] = (__bf16)(v ? fw2[(l*H + k)*H + n] : 0.0f);
    aw1T[(l*256 + n)*FW2_S + k] = (__bf16)(v ? aw1[(l*H + k)*H + n] : 0.0f);
    aw2T[(l*256 + n)*FW2_S + k] = (__bf16)(v ? aw2[(l*H + k)*H + n] : 0.0f);
  }
}

// ---------------------------------------------------------------- interaction (cfconv) kernel
// grid = (64 i-tiles, 8 j-groups), 256 threads (8 waves, 2/SIMD). Full fw2T resident in LDS
// (269,824 B total -- probes the 320KB/workgroup limit). agg accumulated via f32 atomics.
__global__ __launch_bounds__(256) void k_interaction(
    const float* __restrict__ pos, const float* __restrict__ h, float* __restrict__ agg,
    const __bf16* __restrict__ fw1T, const __bf16* __restrict__ fw2T,
    const float* __restrict__ fb1, const float* __restrict__ fb2)
{
  __shared__ alignas(16) __bf16 s_fw1T[256*FW1_S];   //  36,864 B
  __shared__ alignas(16) __bf16 s_fw2T[256*FW2_S];   // 135,168 B (full fw2T)
  __shared__ alignas(16) __bf16 s_G[8*16*FW2_S];     //  67,584 B (per-wave gelu'd GEMM1 out)
  __shared__ alignas(16) __bf16 s_rbf[8*16*FW1_S];   //  18,432 B (per-wave rbf A-tile)
  __shared__ alignas(16) __bf16 s_h[16*H];           //   8,192 B (h of the 16 j atoms, bf16)
  __shared__ float s_dist[256];
  __shared__ float s_fb1[H], s_fb2[H];
  __shared__ float s_pi[16][4], s_pj[16][4];

  const int tid = threadIdx.x, lane = tid & 31, wave = tid >> 5;
  const int itile = blockIdx.x, jg = blockIdx.y;
  const int cl = lane & 15;
  const int mb = (lane & 16) ? 8 : 0;

  // one-time block setup: weights via async DMA to LDS
  {
    const unsigned int l0 = lds_off(s_fw1T);
    for (int e = tid; e < 2304; e += 256)
      async_ld_b128(l0 + (unsigned)e*16u, (unsigned)e*16u, fw1T);
    const unsigned int l1 = lds_off(s_fw2T);
    for (int e = tid; e < 8448; e += 256)
      async_ld_b128(l1 + (unsigned)e*16u, (unsigned)e*16u, fw2T);
    wait_async();
  }
  s_fb1[tid] = fb1[tid];
  s_fb2[tid] = fb2[tid];
  if (tid < 16){
    s_pi[tid][0] = pos[(itile*16 + tid)*3 + 0];
    s_pi[tid][1] = pos[(itile*16 + tid)*3 + 1];
    s_pi[tid][2] = pos[(itile*16 + tid)*3 + 2];
  }

  float aggv[32];
#pragma unroll
  for (int q = 0; q < 32; ++q) aggv[q] = 0.0f;

#pragma unroll 1
  for (int jt = jg*8; jt < jg*8 + 8; ++jt){
    __syncthreads();
    if (tid < 16){
      s_pj[tid][0] = pos[(jt*16 + tid)*3 + 0];
      s_pj[tid][1] = pos[(jt*16 + tid)*3 + 1];
      s_pj[tid][2] = pos[(jt*16 + tid)*3 + 2];
    }
    __syncthreads();
    for (int e = tid; e < 16*H; e += 256)
      s_h[e] = (__bf16)h[(jt*16 + (e >> 8))*H + (e & 255)];
    {
      const int i = tid >> 4, j = tid & 15;
      const float dx = s_pi[i][0]-s_pj[j][0], dy = s_pi[i][1]-s_pj[j][1], dz = s_pi[i][2]-s_pj[j][2];
      s_dist[tid] = sqrtf(dx*dx + dy*dy + dz*dz);
    }
    __syncthreads();

#pragma unroll
    for (int pass = 0; pass < 2; ++pass){
      const int il = pass*8 + wave;                  // this wave's i_local

      // --- radial basis (A of GEMM1), wave-private tile [16 j][64 K] ---
      {
        const float d = s_dist[il*16 + cl];
        const int kbase = (lane >> 4) * 32;
        __bf16* rr = s_rbf + (wave*16 + cl)*FW1_S + kbase;
#pragma unroll 4
        for (int kk = 0; kk < 32; ++kk){
          const int k = kbase + kk;
          const float c = (float)k * (5.0f/49.0f);
          const float v = (k < NG) ? __expf(-200.0f*(d - c)*(d - c)) : 0.0f;
          rr[kk] = (__bf16)v;
        }
      }
      wave_lds_fence();

      // --- GEMM1: [16x64] rbf @ [64x256] fw1 -> gelu -> bf16 G ---
      const __bf16* arow1 = s_rbf + (wave*16 + cl)*FW1_S;
      const v16bf a0 = afrag(arow1, 0,  lane);
      const v16bf a1 = afrag(arow1, 32, lane);
#pragma unroll 4
      for (int nt = 0; nt < 16; ++nt){
        v8f acc = {};
        acc = wmma_bf16(a0, bfrag(s_fw1T + (nt*16 + cl)*FW1_S, 0,  lane), acc);
        acc = wmma_bf16(a1, bfrag(s_fw1T + (nt*16 + cl)*FW1_S, 32, lane), acc);
        const int c = nt*16 + cl;
        const float b = s_fb1[c];
        __bf16* gcol = s_G + wave*16*FW2_S + c;
#pragma unroll
        for (int r = 0; r < 8; ++r)
          gcol[(mb + r)*FW2_S] = (__bf16)gelu_fast(acc[r] + b);
      }
      wave_lds_fence();

      // --- GEMM2: [16x256] G @ [256x256] fw2, fused mask/h/reduce epilogue ---
      const __bf16* arow2 = s_G + (wave*16 + cl)*FW2_S;
      v16bf ga[8];
#pragma unroll
      for (int kt = 0; kt < 8; ++kt) ga[kt] = afrag(arow2, kt*32, lane);

      float msk[8];
#pragma unroll
      for (int r = 0; r < 8; ++r){
        const float dd = s_dist[il*16 + mb + r];
        msk[r] = (dd < 5.0f && dd > 1e-6f) ? 1.0f : 0.0f;
      }

#pragma unroll
      for (int nt = 0; nt < 16; ++nt){
        v8f acc = {};
        const __bf16* brow = s_fw2T + (nt*16 + cl)*FW2_S;
#pragma unroll
        for (int kt = 0; kt < 8; ++kt)
          acc = wmma_bf16(ga[kt], bfrag(brow, kt*32, lane), acc);
        const int c = nt*16 + cl;
        const float b2 = s_fb2[c];
        float s = 0.0f;
#pragma unroll
        for (int r = 0; r < 8; ++r){
          const float hv = (float)s_h[(mb + r)*H + c];
          s += msk[r] * (acc[r] + b2) * hv;        // sum over this lane's 8 j rows
        }
        s += __shfl_xor(s, 16, 32);                // combine the two row halves
        aggv[pass*16 + nt] += s;
      }
    } // pass
  } // jt

#pragma unroll
  for (int pass = 0; pass < 2; ++pass){
    const int i = itile*16 + pass*8 + wave;
    if (lane < 16){
#pragma unroll
      for (int nt = 0; nt < 16; ++nt)
        atomicAdd(&agg[i*H + nt*16 + lane], aggv[pass*16 + nt]);
    }
  }
}

// ---------------------------------------------------------------- atom-wise MLP + residual, re-zero agg
// grid = 64 (16-row M-tiles), 256 threads (8 waves). Full weight matrix resident (152 KB LDS).
__global__ __launch_bounds__(256) void k_atom_mlp(
    float* __restrict__ h, float* __restrict__ agg,
    const __bf16* __restrict__ aw1T, const __bf16* __restrict__ aw2T,
    const float* __restrict__ ab1, const float* __restrict__ ab2)
{
  __shared__ alignas(16) __bf16 s_aT[16*FW2_S];     // agg tile as A (bf16)
  __shared__ alignas(16) __bf16 s_G[16*FW2_S];
  __shared__ alignas(16) __bf16 s_wT[256*FW2_S];    // full 256x256 weight (bf16, transposed)

  const int tid = threadIdx.x, lane = tid & 31, wave = tid >> 5;
  const int row0 = blockIdx.x * 16;
  const int cl = lane & 15;
  const int mb = (lane & 16) ? 8 : 0;

  {
    const unsigned int l0 = lds_off(s_wT);
    for (int e = tid; e < 8448; e += 256)
      async_ld_b128(l0 + (unsigned)e*16u, (unsigned)e*16u, aw1T);
  }
  for (int e = tid; e < 16*H; e += 256){
    const int m = e >> 8, c = e & 255;
    const int gi = (row0 + m)*H + c;
    s_aT[m*FW2_S + c] = (__bf16)agg[gi];
    agg[gi] = 0.0f;                                 // ready for next layer's atomics
  }
  wait_async();
  __syncthreads();

  v16bf a[8];
#pragma unroll
  for (int kt = 0; kt < 8; ++kt) a[kt] = afrag(s_aT + cl*FW2_S, kt*32, lane);

#pragma unroll
  for (int q = 0; q < 2; ++q){
    const int nt = wave*2 + q;
    v8f acc = {};
    const __bf16* brow = s_wT + (nt*16 + cl)*FW2_S;
#pragma unroll
    for (int kt = 0; kt < 8; ++kt) acc = wmma_bf16(a[kt], bfrag(brow, kt*32, lane), acc);
    const int c = nt*16 + cl;
    const float b = ab1[c];
#pragma unroll
    for (int r = 0; r < 8; ++r)
      s_G[(mb + r)*FW2_S + c] = (__bf16)gelu_exact(acc[r] + b);
  }
  __syncthreads();                                  // GEMM1 done; s_wT free, s_G complete

  v16bf ga[8];
#pragma unroll
  for (int kt = 0; kt < 8; ++kt) ga[kt] = afrag(s_G + cl*FW2_S, kt*32, lane);

  {
    const unsigned int l0 = lds_off(s_wT);
    for (int e = tid; e < 8448; e += 256)
      async_ld_b128(l0 + (unsigned)e*16u, (unsigned)e*16u, aw2T);
    wait_async();
  }
  __syncthreads();

#pragma unroll
  for (int q = 0; q < 2; ++q){
    const int nt = wave*2 + q;
    v8f acc = {};
    const __bf16* brow = s_wT + (nt*16 + cl)*FW2_S;
#pragma unroll
    for (int kt = 0; kt < 8; ++kt) acc = wmma_bf16(ga[kt], bfrag(brow, kt*32, lane), acc);
    const int c = nt*16 + cl;
    const float b = ab2[c];
#pragma unroll
    for (int r = 0; r < 8; ++r){
      const int gi = (row0 + mb + r)*H + c;
      h[gi] = h[gi] + acc[r] + b;                   // residual, in-place (rows block-owned)
    }
  }
}

// ---------------------------------------------------------------- pool + projection head + layernorm
__global__ __launch_bounds__(256) void k_head(
    const float* __restrict__ h, const float* __restrict__ pw1, const float* __restrict__ pb1,
    const float* __restrict__ pw2, const float* __restrict__ pb2,
    const float* __restrict__ lng, const float* __restrict__ lnb, float* __restrict__ out)
{
  __shared__ float sp[NB*H];
  __shared__ float st[NB*H];
  __shared__ float sx[NB*LDIM];
  __shared__ float smu[NB], srs[NB];
  const int tid = threadIdx.x;

  for (int e = tid; e < NB*H; e += 256){
    const int b = e >> 8, c = e & 255;
    float s = 0.0f;
    for (int a2 = 0; a2 < 64; ++a2) s += h[(b*64 + a2)*H + c];
    sp[e] = s;
  }
  __syncthreads();
  for (int e = tid; e < NB*H; e += 256){
    const int b = e >> 8, c = e & 255;
    float s = pb1[c];
    for (int k = 0; k < H; ++k) s += sp[b*H + k] * pw1[k*H + c];
    st[e] = gelu_exact(s);
  }
  __syncthreads();
  for (int e = tid; e < NB*LDIM; e += 256){
    const int b = e >> 9, l2 = e & 511;
    float s = pb2[l2];
    for (int k = 0; k < H; ++k) s += st[b*H + k] * pw2[k*LDIM + l2];
    sx[e] = s;
  }
  __syncthreads();
  if (tid < NB){
    float mu = 0.0f;
    for (int l2 = 0; l2 < LDIM; ++l2) mu += sx[tid*LDIM + l2];
    mu *= (1.0f/LDIM);
    float v = 0.0f;
    for (int l2 = 0; l2 < LDIM; ++l2){ const float d = sx[tid*LDIM + l2] - mu; v += d*d; }
    v *= (1.0f/LDIM);
    smu[tid] = mu; srs[tid] = rsqrtf(v + 1e-5f);
  }
  __syncthreads();
  for (int e = tid; e < NB*LDIM; e += 256){
    const int b = e >> 9, l2 = e & 511;
    out[e] = (sx[e] - smu[b]) * srs[b] * lng[l2] + lnb[l2];
  }
}

// ---------------------------------------------------------------- launch
extern "C" void kernel_launch(void* const* d_in, const int* in_sizes, int n_in,
                              void* d_out, int out_size, void* d_ws, size_t ws_size,
                              hipStream_t stream)
{
  const int*   z   = (const int*)  d_in[0];
  const float* pos = (const float*)d_in[1];
  const float* emb = (const float*)d_in[3];
  const float* fw1 = (const float*)d_in[4];
  const float* fb1 = (const float*)d_in[5];
  const float* fw2 = (const float*)d_in[6];
  const float* fb2 = (const float*)d_in[7];
  const float* aw1 = (const float*)d_in[8];
  const float* ab1 = (const float*)d_in[9];
  const float* aw2 = (const float*)d_in[10];
  const float* ab2 = (const float*)d_in[11];
  const float* pw1 = (const float*)d_in[12];
  const float* pb1 = (const float*)d_in[13];
  const float* pw2 = (const float*)d_in[14];
  const float* pb2 = (const float*)d_in[15];
  const float* lng = (const float*)d_in[16];
  const float* lnb = (const float*)d_in[17];

  char* ws = (char*)d_ws;
  float*  h    = (float*)(ws);
  float*  agg  = (float*)(ws + (1u<<20));
  __bf16* fw1T = (__bf16*)(ws + 2u*(1u<<20));
  __bf16* fw2T = (__bf16*)(ws + 2u*(1u<<20) + 110592u);
  __bf16* aw1T = (__bf16*)(ws + 2u*(1u<<20) + 110592u + 405504u);
  __bf16* aw2T = (__bf16*)(ws + 2u*(1u<<20) + 110592u + 2u*405504u);

  k_init<<<N_ATOMS, H, 0, stream>>>(z, emb, h, agg);
  k_pack<<<NI*256, 128, 0, stream>>>(fw1, fw2, aw1, aw2, fw1T, fw2T, aw1T, aw2T);
  for (int l = 0; l < NI; ++l){
    k_interaction<<<dim3(64, 8), 256, 0, stream>>>(pos, h, agg,
        fw1T + (size_t)l*256*FW1_S, fw2T + (size_t)l*256*FW2_S, fb1 + l*H, fb2 + l*H);
    k_atom_mlp<<<64, 256, 0, stream>>>(h, agg,
        aw1T + (size_t)l*256*FW2_S, aw2T + (size_t)l*256*FW2_S, ab1 + l*H, ab2 + l*H);
  }
  k_head<<<1, 256, 0, stream>>>(h, pw1, pb1, pw2, pb2, lng, lnb, (float*)d_out);
}
